// ConditionalTransformer_24335284699625
// MI455X (gfx1250) — matrix operand order repaired and checked by reading, not compiled, not run on hardware
//
#include <hip/hip_runtime.h>
#include <math.h>

#define DEVFN __device__ __forceinline__

typedef __attribute__((ext_vector_type(16))) _Float16 v16h;
typedef __attribute__((ext_vector_type(8)))  _Float16 v8h;
typedef __attribute__((ext_vector_type(8)))  float    v8f;
typedef __attribute__((ext_vector_type(4)))  float    f32x4;

// ---------------- problem dims (from setup_inputs) ----------------
#define BB     4
#define TT     32
#define GGC    4
#define NOBS   2
#define NSCN   2048
#define EE     120
#define HH     4
#define HD     30
#define FEATC  60
#define GCONDC 512
#define CONDD  632          // GCOND + E
#define L_OGP  (NOBS*GGC)   // 8
#define L_SCN  (NOBS*NSCN)  // 4096
#define L_SP   (TT*GGC)     // 128

static inline int ceil_div(int a, int b) { return (a + b - 1) / b; }

// ---------------- small load/convert helpers ----------------
DEVFN f32x4 load4(const float* p) {  // dword-aligned-safe vector load
  f32x4 v;
  __builtin_memcpy(&v, p, 16);
  return v;
}
// convert 16 consecutive f32 -> 16 f16 in LDS (two b128 stores)
DEVFN void cvt16(_Float16* dst, const float* src) {
  f32x4 a = load4(src), b = load4(src + 4), c = load4(src + 8), d = load4(src + 12);
  v8h h0 = {(_Float16)a.x, (_Float16)a.y, (_Float16)a.z, (_Float16)a.w,
            (_Float16)b.x, (_Float16)b.y, (_Float16)b.z, (_Float16)b.w};
  v8h h1 = {(_Float16)c.x, (_Float16)c.y, (_Float16)c.z, (_Float16)c.w,
            (_Float16)d.x, (_Float16)d.y, (_Float16)d.z, (_Float16)d.w};
  *(v8h*)(dst) = h0;
  *(v8h*)(dst + 8) = h1;
}

// ---------------- WMMA fragment gathers ----------------
// A operand tile: 16x32 row-major [m][k]
DEVFN v16h frag_a(const _Float16* As, int lane) {
  v16h a;
  const int m = lane & 15, hi = lane >> 4;
#pragma unroll
  for (int v = 0; v < 8; ++v) {
    const int k0 = ((v >> 2) << 4) + (hi << 3) + ((v & 3) << 1);
    a[2 * v]     = As[m * 32 + k0];
    a[2 * v + 1] = As[m * 32 + k0 + 1];
  }
  return a;
}
// B operand from N-major tile [n][k] (ld=32), column group nOff..nOff+15
DEVFN v16h frag_b_nk(const _Float16* Bs, int lane, int nOff) {
  v16h b;
  const int n = (lane & 15) + nOff, hi = lane >> 4;
#pragma unroll
  for (int v = 0; v < 16; ++v) b[v] = Bs[n * 32 + ((v >> 3) << 4) + (hi << 3) + (v & 7)];
  return b;
}
// B operand from K-major tile [k][n] (ld=32), column group nOff..nOff+15
DEVFN v16h frag_b_kn(const _Float16* Bs, int lane, int nOff) {
  v16h b;
  const int n = (lane & 15) + nOff, hi = lane >> 4;
#pragma unroll
  for (int v = 0; v < 16; ++v) b[v] = Bs[(((v >> 3) << 4) + (hi << 3) + (v & 7)) * 32 + n];
  return b;
}

DEVFN float mishf(float x) {
  float sp = (x > 20.f) ? x : log1pf(expf(x));
  return x * tanhf(sp);
}

// ---------------- GEMM: tokens x weight^T ----------------
// C[M,N] = act(alpha*(A[M,K] @ W[N,K]^T + bias[N]));  wave tile = 16M x 32N
__global__ __launch_bounds__(128) void k_gemm_tok(const float* __restrict__ A,
                                                  const float* __restrict__ W,
                                                  const float* __restrict__ bias,
                                                  float* __restrict__ C, int M, int N,
                                                  int K, float alpha, int relu) {
  __shared__ alignas(16) _Float16 sA[4][16 * 32];
  __shared__ alignas(16) _Float16 sB[4][32 * 32];
  const int lane  = threadIdx.x & 31;
  const int wave  = threadIdx.x >> 5;
  const int nBase = blockIdx.x * 32;
  const int mBase = (blockIdx.y * 4 + wave) * 16;
  const int rA = lane >> 1, cA = (lane & 1) << 4;  // A fill: 2 lanes per row
  v8f acc0 = {}, acc1 = {};
  for (int k0 = 0; k0 < K; k0 += 32) {
    // ---- A tile 16x32 [m][k] ----
    if (mBase + 16 <= M && k0 + 32 <= K) {
      cvt16(&sA[wave][rA * 32 + cA], &A[(size_t)(mBase + rA) * K + k0 + cA]);
    } else {
#pragma unroll
      for (int j = 0; j < 16; ++j) {
        const int gm = mBase + rA, gk = k0 + cA + j;
        sA[wave][rA * 32 + cA + j] =
            (gm < M && gk < K) ? (_Float16)A[(size_t)gm * K + gk] : (_Float16)0.f;
      }
    }
    // ---- B tile 32n x 32k [n][k] ---- (one row per lane)
    if (nBase + 32 <= N && k0 + 32 <= K) {
      const float* wp = &W[(size_t)(nBase + lane) * K + k0];
      cvt16(&sB[wave][lane * 32], wp);
      cvt16(&sB[wave][lane * 32 + 16], wp + 16);
    } else {
      const int gn = nBase + lane;
#pragma unroll
      for (int j = 0; j < 32; ++j) {
        const int gk = k0 + j;
        sB[wave][lane * 32 + j] =
            (gn < N && gk < K) ? (_Float16)W[(size_t)gn * K + gk] : (_Float16)0.f;
      }
    }
    __syncthreads();
    v16h a  = frag_a(sA[wave], lane);
    v16h b0 = frag_b_nk(sB[wave], lane, 0);
    v16h b1 = frag_b_nk(sB[wave], lane, 16);
    acc0 = __builtin_amdgcn_wmma_f32_16x16x32_f16(false, a, false, b0, (short)0, acc0,
                                                  false, false);
    acc1 = __builtin_amdgcn_wmma_f32_16x16x32_f16(false, a, false, b1, (short)0, acc1,
                                                  false, false);
    __syncthreads();
  }
  const int hi = lane >> 4, n = lane & 15;
#pragma unroll
  for (int h = 0; h < 2; ++h) {
    const int gn = nBase + h * 16 + n;
    if (gn < N) {
      const float bv = bias ? bias[gn] : 0.f;
      const v8f& acc = h ? acc1 : acc0;
#pragma unroll
      for (int v = 0; v < 8; ++v) {
        const int m = mBase + v + 8 * hi;
        if (m < M) {
          float val = alpha * (acc[v] + bv);
          if (relu) val = fmaxf(val, 0.f);
          C[(size_t)m * N + gn] = val;
        }
      }
    }
  }
}

// ---------------- implicit-GEMM conv1d (per batch) ----------------
// Y[b,co,l] = sum_{ci,dk} W[co,ci,dk] * X[b,ci,l+dk-pad] + bias[co]
// wave tile = 16co x 32l; im2col tile staged K-major [k][l] for contiguous loads
__global__ __launch_bounds__(128) void k_gemm_conv(const float* __restrict__ Wt,
                                                   const float* __restrict__ bias,
                                                   const float* __restrict__ X,
                                                   float* __restrict__ Y, int Cin,
                                                   int Cout, int L, int KW, int pad) {
  __shared__ alignas(16) _Float16 sW[4][16 * 32];
  __shared__ alignas(16) _Float16 sX[4][32 * 32];
  const int lane   = threadIdx.x & 31;
  const int wave   = threadIdx.x >> 5;
  const int lBase  = blockIdx.x * 32;
  const int coBase = (blockIdx.y * 4 + wave) * 16;
  const int b      = blockIdx.z;
  const float* Xb  = X + (size_t)b * Cin * L;
  const int K      = Cin * KW;
  const int rA = lane >> 1, cA = (lane & 1) << 4;
  v8f acc0 = {}, acc1 = {};
  for (int k0 = 0; k0 < K; k0 += 32) {
    // ---- W tile 16x32 [co][k] ----
    if (coBase + 16 <= Cout && k0 + 32 <= K) {
      cvt16(&sW[wave][rA * 32 + cA], &Wt[(size_t)(coBase + rA) * K + k0 + cA]);
    } else {
#pragma unroll
      for (int j = 0; j < 16; ++j) {
        const int gc = coBase + rA, gk = k0 + cA + j;
        sW[wave][rA * 32 + cA + j] =
            (gc < Cout && gk < K) ? (_Float16)Wt[(size_t)gc * K + gk] : (_Float16)0.f;
      }
    }
    // ---- X tile 32k x 32l [k][l] ---- (one k-row of 32 contiguous l's per lane)
    {
      const int gk = k0 + lane;
      const int ci = gk / KW, dk = gk % KW;
      const int l0 = lBase + dk - pad;
      if (gk < K && l0 >= 0 && l0 + 32 <= L) {
        const float* xp = Xb + (size_t)ci * L + l0;
        cvt16(&sX[wave][lane * 32], xp);
        cvt16(&sX[wave][lane * 32 + 16], xp + 16);
      } else {
#pragma unroll
        for (int j = 0; j < 32; ++j) {
          const int l = l0 + j;
          sX[wave][lane * 32 + j] = (gk < K && l >= 0 && l < L)
                                        ? (_Float16)Xb[(size_t)ci * L + l]
                                        : (_Float16)0.f;
        }
      }
    }
    __syncthreads();
    v16h a  = frag_a(sW[wave], lane);
    v16h b0 = frag_b_kn(sX[wave], lane, 0);
    v16h b1 = frag_b_kn(sX[wave], lane, 16);
    acc0 = __builtin_amdgcn_wmma_f32_16x16x32_f16(false, a, false, b0, (short)0, acc0,
                                                  false, false);
    acc1 = __builtin_amdgcn_wmma_f32_16x16x32_f16(false, a, false, b1, (short)0, acc1,
                                                  false, false);
    __syncthreads();
  }
  const int hi = lane >> 4, n = lane & 15;
#pragma unroll
  for (int h = 0; h < 2; ++h) {
    const int l = lBase + h * 16 + n;
    if (l < L) {
      const v8f& acc = h ? acc1 : acc0;
#pragma unroll
      for (int v = 0; v < 8; ++v) {
        const int co = coBase + v + 8 * hi;
        if (co < Cout)
          Y[((size_t)b * Cout + co) * L + l] = acc[v] + (bias ? bias[co] : 0.f);
      }
    }
  }
}

// ---------------- per-sample mean/var over (C,L) ----------------
__global__ void k_meanvar(const float* __restrict__ Y, int C, int L,
                          float* __restrict__ stat) {
  __shared__ float ss[256];
  __shared__ float s2[256];
  const int b = blockIdx.x, t = threadIdx.x;
  const size_t n = (size_t)C * L;
  const float* p = Y + (size_t)b * n;
  double a = 0.0, q = 0.0;
  for (size_t i = t; i < n; i += 256) {
    float v = p[i];
    a += v;
    q += (double)v * v;
  }
  ss[t] = (float)a;
  s2[t] = (float)q;
  __syncthreads();
  for (int o = 128; o > 0; o >>= 1) {
    if (t < o) { ss[t] += ss[t + o]; s2[t] += s2[t + o]; }
    __syncthreads();
  }
  if (t == 0) {
    float mu  = ss[0] / (float)n;
    float var = s2[0] / (float)n - mu * mu;
    stat[b * 2]     = mu;
    stat[b * 2 + 1] = rsqrtf(var + 1e-5f);
  }
}

// ---------------- cblock epilogue: norm + affine + mish (+FiLM) (+residual) ----
__global__ void k_cblock_ep(const float* __restrict__ Y, float* __restrict__ Out,
                            const float* __restrict__ stat, const float* __restrict__ g,
                            const float* __restrict__ be, const float* __restrict__ film,
                            const float* __restrict__ add, int C, int L) {
  const size_t tot = (size_t)BB * C * L;
  size_t idx = (size_t)blockIdx.x * blockDim.x + threadIdx.x;
  if (idx >= tot) return;
  const int c = (int)((idx / L) % C);
  const int b = (int)(idx / ((size_t)C * L));
  const float mu = stat[b * 2], rs = stat[b * 2 + 1];
  float v = (Y[idx] - mu) * rs * g[c] + be[c];
  v = mishf(v);
  if (film) v = film[(size_t)b * 2 * C + c] * v + film[(size_t)b * 2 * C + C + c];
  if (add) v += add[idx];
  Out[idx] = v;
}

// ---------------- layout kernels ----------------
__global__ void k_cl_to_lbc(const float* __restrict__ X, float* __restrict__ Y, int C,
                            int L) {
  size_t idx = (size_t)blockIdx.x * blockDim.x + threadIdx.x;
  if (idx >= (size_t)BB * C * L) return;
  const int l = (int)(idx % L);
  const int c = (int)((idx / L) % C);
  const int b = (int)(idx / ((size_t)C * L));
  Y[((size_t)l * BB + b) * C + c] = X[idx];
}

__global__ void k_lbc_to_cl(const float* __restrict__ X, float* __restrict__ Y, int C,
                            int L) {
  size_t idx = (size_t)blockIdx.x * blockDim.x + threadIdx.x;
  if (idx >= (size_t)BB * C * L) return;
  const int l = (int)(idx % L);
  const int c = (int)((idx / L) % C);
  const int b = (int)(idx / ((size_t)C * L));
  Y[idx] = X[((size_t)l * BB + b) * C + c];
}

__global__ void k_sinpos(const int* __restrict__ iv, int useIota, int n,
                         float* __restrict__ out) {
  int idx = blockIdx.x * blockDim.x + threadIdx.x;
  if (idx >= n * EE) return;
  const int r = idx / EE, e = idx % EE;
  const float v = useIota ? (float)r : (float)iv[r];
  const int i = (e < 60) ? e : e - 60;
  const float f = expf(-(float)i * logf(10000.f) / 59.f);
  const float ang = v * f;
  out[idx] = (e < 60) ? sinf(ang) : cosf(ang);
}

__global__ void k_gfeat(const float* __restrict__ temb, const float* __restrict__ gc,
                        float* __restrict__ mishg) {
  int idx = blockIdx.x * blockDim.x + threadIdx.x;
  if (idx >= BB * CONDD) return;
  const int b = idx / CONDD, e = idx % CONDD;
  const float x = (e < EE) ? temb[b * EE + e] : gc[(size_t)b * GCONDC + (e - EE)];
  mishg[idx] = mishf(x);
}

__global__ void k_build_ogp_x(const float* __restrict__ tok, const float* __restrict__ obs,
                              float* __restrict__ X) {
  int idx = blockIdx.x * blockDim.x + threadIdx.x;
  if (idx >= BB * 240 * L_OGP) return;
  const int l = idx % L_OGP;
  const int c = (idx / L_OGP) % 240;
  const int b = idx / (240 * L_OGP);
  const int o = l >> 2, g = l & 3;
  X[idx] = (c < EE) ? tok[(((size_t)b * 2 + o) * 4 + g) * EE + c]
                    : obs[o * EE + (c - EE)];
}

__global__ void k_build_scn_x(const float* __restrict__ feat,
                              const float* __restrict__ obs, float* __restrict__ X) {
  size_t idx = (size_t)blockIdx.x * blockDim.x + threadIdx.x;
  if (idx >= (size_t)BB * 180 * L_SCN) return;
  const int l = (int)(idx % L_SCN);
  const int c = (int)((idx / L_SCN) % 180);
  const int b = (int)(idx / ((size_t)180 * L_SCN));
  const int o = l >> 11, nn = l & 2047;
  X[idx] = (c < FEATC) ? feat[(((size_t)b * 2 + o) * NSCN + nn) * FEATC + c]
                       : obs[o * EE + (c - FEATC)];
}

__global__ void k_build_sp_x(const float* __restrict__ smp, float* __restrict__ X) {
  int idx = blockIdx.x * blockDim.x + threadIdx.x;
  if (idx >= BB * 3 * L_SP) return;
  const int l = idx % L_SP;
  const int a = (idx / L_SP) % 3;
  const int b = idx / (3 * L_SP);
  const int t = l >> 2, g = l & 3;
  X[idx] = smp[((size_t)b * TT + t) * 12 + g * 3 + a];
}

__global__ void k_build_spcat(const float* __restrict__ spe, const float* __restrict__ act,
                              float* __restrict__ X) {
  int idx = blockIdx.x * blockDim.x + threadIdx.x;
  if (idx >= BB * 240 * L_SP) return;
  const int l = idx % L_SP;
  const int c = (idx / L_SP) % 240;
  const int b = idx / (240 * L_SP);
  const int t = l >> 2;
  X[idx] = (c < EE) ? spe[((size_t)b * EE + c) * L_SP + l] : act[t * EE + (c - EE)];
}

__global__ void k_out_final(const float* __restrict__ Y, float* __restrict__ out) {
  int idx = blockIdx.x * blockDim.x + threadIdx.x;
  if (idx >= BB * TT * 12) return;
  const int j = idx % 12;
  const int t = (idx / 12) % TT;
  const int b = idx / (12 * TT);
  const int g = j / 3, a = j % 3;
  out[idx] = Y[((size_t)b * 3 + a) * L_SP + t * 4 + g];
}

// ---------------- rotary ----------------
__global__ void k_rope(const float* __restrict__ pts, float* __restrict__ pe, int L) {
  size_t idx = (size_t)blockIdx.x * blockDim.x + threadIdx.x;
  if (idx >= (size_t)BB * L * 60) return;
  const int t = (int)(idx % 60);
  const int l = (int)((idx / 60) % L);
  const int b = (int)(idx / ((size_t)60 * L));
  const int a = t / 20, j = t % 20;
  const float x = pts[((size_t)b * L + l) * 3 + a];
  const float f = expf(-(float)(2 * j) * logf(10000.f) / 40.f);
  const float cv = cosf(x * f), sv = sinf(x * f);
  const size_t base = (((size_t)b * L + l) * EE + a * 40 + 2 * j) * 2;
  pe[base] = cv; pe[base + 1] = sv; pe[base + 2] = cv; pe[base + 3] = sv;
}

__global__ void k_rotary(float* __restrict__ Xq, const float* __restrict__ pe, int L) {
  size_t idx = (size_t)blockIdx.x * blockDim.x + threadIdx.x;
  if (idx >= (size_t)L * BB * 60) return;
  const int p = (int)(idx % 60);
  const int b = (int)((idx / 60) % BB);
  const int l = (int)(idx / (60 * BB));
  const int e = 2 * p;
  const size_t xi = ((size_t)l * BB + b) * EE + e;
  const float x0 = Xq[xi], x1 = Xq[xi + 1];
  const size_t pb = (((size_t)b * L + l) * EE + e) * 2;
  Xq[xi]     = x0 * pe[pb] - x1 * pe[pb + 1];
  Xq[xi + 1] = x1 * pe[pb + 2] + x0 * pe[pb + 3];
}

// ---------------- residual add + layernorm (E=120) ----------------
__global__ void k_add_ln(const float* __restrict__ A, const float* __restrict__ Bv,
                         const float* __restrict__ g, const float* __restrict__ be,
                         float* __restrict__ Out) {
  __shared__ float sh[128];
  __shared__ float sh2[128];
  const int r = blockIdx.x, t = threadIdx.x;
  float v = 0.f;
  if (t < EE) v = A[(size_t)r * EE + t] + Bv[(size_t)r * EE + t];
  sh[t] = v; sh2[t] = v * v;
  __syncthreads();
  for (int o = 64; o > 0; o >>= 1) {
    if (t < o) { sh[t] += sh[t + o]; sh2[t] += sh2[t + o]; }
    __syncthreads();
  }
  const float mu = sh[0] / (float)EE;
  const float var = sh2[0] / (float)EE - mu * mu;
  const float rs = rsqrtf(var + 1e-5f);
  if (t < EE) Out[(size_t)r * EE + t] = (v - mu) * rs * g[t] + be[t];
}

// ---------------- softmax over rows of length Lk ----------------
__global__ void k_softmax(float* __restrict__ S, int Lk) {
  __shared__ float sh[256];
  const int r = blockIdx.x, t = threadIdx.x;
  float* row = S + (size_t)r * Lk;
  float m = -1e30f;
  for (int i = t; i < Lk; i += 256) m = fmaxf(m, row[i]);
  sh[t] = m; __syncthreads();
  for (int o = 128; o > 0; o >>= 1) { if (t < o) sh[t] = fmaxf(sh[t], sh[t + o]); __syncthreads(); }
  m = sh[0]; __syncthreads();
  float s = 0.f;
  for (int i = t; i < Lk; i += 256) { float e = expf(row[i] - m); row[i] = e; s += e; }
  sh[t] = s; __syncthreads();
  for (int o = 128; o > 0; o >>= 1) { if (t < o) sh[t] += sh[t + o]; __syncthreads(); }
  const float inv = 1.f / sh[0];
  for (int i = t; i < Lk; i += 256) row[i] *= inv;
}

// ---------------- attention score / PV WMMA kernels ----------------
__global__ __launch_bounds__(32) void k_attn_qk(const float* __restrict__ Q,
                                                const float* __restrict__ Kk,
                                                float* __restrict__ S, int Lq, int Lk) {
  const int lane = threadIdx.x;
  const int lkBase = blockIdx.x * 16, lqBase = blockIdx.y * 16;
  const int bh = blockIdx.z, b = bh >> 2, h = bh & 3;
  const int m = lane & 15, hi = lane >> 4;
  v16h a, bb;
#pragma unroll
  for (int v = 0; v < 8; ++v) {
    const int k0 = ((v >> 2) << 4) + (hi << 3) + ((v & 3) << 1);
    const size_t qb = ((size_t)(lqBase + m) * BB + b) * EE + h * HD;
    a[2 * v]     = (k0 < HD)     ? (_Float16)Q[qb + k0]     : (_Float16)0.f;
    a[2 * v + 1] = (k0 + 1 < HD) ? (_Float16)Q[qb + k0 + 1] : (_Float16)0.f;
  }
  const int lk = lkBase + m;
#pragma unroll
  for (int v = 0; v < 16; ++v) {
    const int d = ((v >> 3) << 4) + (hi << 3) + (v & 7);
    bb[v] = (d < HD && lk < Lk)
                ? (_Float16)Kk[((size_t)lk * BB + b) * EE + h * HD + d]
                : (_Float16)0.f;
  }
  v8f acc = {};
  acc = __builtin_amdgcn_wmma_f32_16x16x32_f16(false, a, false, bb, (short)0, acc,
                                               false, false);
  if (lk < Lk) {
#pragma unroll
    for (int v = 0; v < 8; ++v) {
      const int lq = lqBase + v + 8 * hi;
      S[((size_t)bh * Lq + lq) * Lk + lk] = acc[v];
    }
  }
}

__global__ __launch_bounds__(32) void k_attn_av(const float* __restrict__ S,
                                                const float* __restrict__ V,
                                                float* __restrict__ O, int Lq, int Lk) {
  const int lane = threadIdx.x;
  const int dBase = blockIdx.x * 16, lqBase = blockIdx.y * 16;
  const int bh = blockIdx.z, b = bh >> 2, h = bh & 3;
  const int m = lane & 15, hi = lane >> 4;
  v8f acc = {};
  for (int kb = 0; kb < Lk; kb += 32) {
    v16h a, bb;
    const float* row = S + ((size_t)bh * Lq + lqBase + m) * Lk + kb;
    if (kb + 32 <= Lk) {
      // A layout per lane = two contiguous 8-float chunks: [hi*8..+7], [16+hi*8..+7]
      f32x4 p0 = load4(row + (hi << 3));
      f32x4 p1 = load4(row + (hi << 3) + 4);
      f32x4 p2 = load4(row + 16 + (hi << 3));
      f32x4 p3 = load4(row + 16 + (hi << 3) + 4);
      a = v16h{(_Float16)p0.x, (_Float16)p0.y, (_Float16)p0.z, (_Float16)p0.w,
               (_Float16)p1.x, (_Float16)p1.y, (_Float16)p1.z, (_Float16)p1.w,
               (_Float16)p2.x, (_Float16)p2.y, (_Float16)p2.z, (_Float16)p2.w,
               (_Float16)p3.x, (_Float16)p3.y, (_Float16)p3.z, (_Float16)p3.w};
    } else {
#pragma unroll
      for (int v = 0; v < 8; ++v) {
        const int k0 = ((v >> 2) << 4) + (hi << 3) + ((v & 3) << 1);
        a[2 * v]     = (kb + k0 < Lk)     ? (_Float16)row[k0]     : (_Float16)0.f;
        a[2 * v + 1] = (kb + k0 + 1 < Lk) ? (_Float16)row[k0 + 1] : (_Float16)0.f;
      }
    }
    const int d = dBase + m;
#pragma unroll
    for (int v = 0; v < 16; ++v) {
      const int lk = kb + ((v >> 3) << 4) + (hi << 3) + (v & 7);
      bb[v] = (lk < Lk && d < HD)
                  ? (_Float16)V[((size_t)lk * BB + b) * EE + h * HD + d]
                  : (_Float16)0.f;
    }
    acc = __builtin_amdgcn_wmma_f32_16x16x32_f16(false, a, false, bb, (short)0, acc,
                                                 false, false);
  }
  const int d = dBase + m;
  if (d < HD) {
#pragma unroll
    for (int v = 0; v < 8; ++v) {
      const int lq = lqBase + v + 8 * hi;
      O[((size_t)lq * BB + b) * EE + h * HD + d] = acc[v];
    }
  }
}

// =================== host-side orchestration ===================
struct AttnL {
  const float *ff1b, *ff1w, *ff2b, *ff2w, *inb, *inw, *ln1b, *ln1g, *ln2b, *ln2g,
      *outb, *outw;
};
struct Film {
  const float *b0b, *b0w, *b0gb, *b0gg, *b1b, *b1w, *b1gb, *b1gg, *cb, *cw, *rb, *rw;
  int cin, cout;
};

static void gemm_tok(hipStream_t st, const float* A, const float* W, const float* bias,
                     float* C, int M, int N, int K, float alpha = 1.f, int relu = 0) {
  dim3 g(ceil_div(N, 32), ceil_div(ceil_div(M, 16), 4));
  k_gemm_tok<<<g, 128, 0, st>>>(A, W, bias, C, M, N, K, alpha, relu);
}
static void conv(hipStream_t st, const float* Wt, const float* bias, const float* X,
                 float* Y, int Cin, int Cout, int L, int KW, int pad) {
  dim3 g(ceil_div(L, 32), ceil_div(ceil_div(Cout, 16), 4), BB);
  k_gemm_conv<<<g, 128, 0, st>>>(Wt, bias, X, Y, Cin, Cout, L, KW, pad);
}

static void run_film(hipStream_t st, const Film& f, const float* X, float* Out, float* y0,
                     float* y1, const float* mishg, float* ebuf, float* stat, int L) {
  const int C = f.cout;
  const int tot = BB * C * L;
  conv(st, f.b0w, f.b0b, X, y0, f.cin, C, L, 3, 1);
  k_meanvar<<<BB, 256, 0, st>>>(y0, C, L, stat);
  gemm_tok(st, mishg, f.cw, f.cb, ebuf, BB, 2 * C, CONDD);
  k_cblock_ep<<<ceil_div(tot, 256), 256, 0, st>>>(y0, y0, stat, f.b0gg, f.b0gb, ebuf,
                                                  nullptr, C, L);
  conv(st, f.b1w, f.b1b, y0, y1, C, C, L, 3, 1);
  k_meanvar<<<BB, 256, 0, st>>>(y1, C, L, stat);
  const float* addp;
  if (f.rw) { conv(st, f.rw, f.rb, X, y0, f.cin, C, L, 1, 0); addp = y0; }
  else addp = X;
  k_cblock_ep<<<ceil_div(tot, 256), 256, 0, st>>>(y1, Out, stat, f.b1gg, f.b1gb, nullptr,
                                                  addp, C, L);
}

static void run_fcrb(hipStream_t st, const Film* f, const float* Xin, float* Xout,
                     float* s0, float* s1, float* s2, const float* mishg, float* ebuf,
                     float* stat, int L) {
  run_film(st, f[0], Xin, s1, s0, s1, mishg, ebuf, stat, L);
  run_film(st, f[1], s1, s2, s0, s2, mishg, ebuf, stat, L);
  run_film(st, f[2], s2, Xout, s0, s1, mishg, ebuf, stat, L);
}

static void run_attn(hipStream_t st, const AttnL& p, const float* q, const float* v,
                     const float* peQ, const float* peV, int Lq, int Lk, float* qq,
                     float* kk, float* vv, float* oo, float* x1, float* f1, float* f2,
                     float* S, float* qout) {
  const float scale = 0.1825741858f;  // 30^-0.5
  gemm_tok(st, q, p.inw, p.inb, qq, Lq * BB, EE, EE, scale, 0);
  gemm_tok(st, v, p.inw + EE * EE, p.inb + EE, kk, Lk * BB, EE, EE, 1.f, 0);
  gemm_tok(st, v, p.inw + 2 * EE * EE, p.inb + 2 * EE, vv, Lk * BB, EE, EE, 1.f, 0);
  k_rotary<<<ceil_div(Lq * BB * 60, 256), 256, 0, st>>>(qq, peQ, Lq);
  k_rotary<<<ceil_div(Lk * BB * 60, 256), 256, 0, st>>>(kk, peV, Lk);
  dim3 gqk(ceil_div(Lk, 16), Lq / 16, BB * HH);
  k_attn_qk<<<gqk, 32, 0, st>>>(qq, kk, S, Lq, Lk);
  k_softmax<<<BB * HH * Lq, 256, 0, st>>>(S, Lk);
  dim3 gav(2, Lq / 16, BB * HH);
  k_attn_av<<<gav, 32, 0, st>>>(S, vv, oo, Lq, Lk);
  gemm_tok(st, oo, p.outw, p.outb, f2, Lq * BB, EE, EE);
  k_add_ln<<<Lq * BB, 128, 0, st>>>(q, f2, p.ln1g, p.ln1b, x1);
  gemm_tok(st, x1, p.ff1w, p.ff1b, f1, Lq * BB, 4 * EE, EE, 1.f, 1);
  gemm_tok(st, f1, p.ff2w, p.ff2b, f2, Lq * BB, EE, 4 * EE);
  k_add_ln<<<Lq * BB, 128, 0, st>>>(x1, f2, p.ln2g, p.ln2b, qout);
}

// ---- param pytree parsing: JAX tree order (sorted dict keys; None res -> no leaf) ----
static void parse_attn(void* const* din, int& c, AttnL* L3) {
  for (int i = 0; i < 3; ++i) {
    AttnL& a = L3[i];
    a.ff1b = (const float*)din[c++]; a.ff1w = (const float*)din[c++];
    a.ff2b = (const float*)din[c++]; a.ff2w = (const float*)din[c++];
    a.inb  = (const float*)din[c++]; a.inw  = (const float*)din[c++];
    a.ln1b = (const float*)din[c++]; a.ln1g = (const float*)din[c++];
    a.ln2b = (const float*)din[c++]; a.ln2g = (const float*)din[c++];
    a.outb = (const float*)din[c++]; a.outw = (const float*)din[c++];
  }
}
static void parse_fcrb(void* const* din, int& c, Film* f3, int cin, int cout) {
  const int cs[4] = {cin, 4 * cout, 4 * cout, cout};
  for (int i = 0; i < 3; ++i) {
    Film& f = f3[i];
    f.cin = cs[i]; f.cout = cs[i + 1];
    f.b0b = (const float*)din[c++]; f.b0w = (const float*)din[c++];
    f.b0gb = (const float*)din[c++]; f.b0gg = (const float*)din[c++];
    f.b1b = (const float*)din[c++]; f.b1w = (const float*)din[c++];
    f.b1gb = (const float*)din[c++]; f.b1gg = (const float*)din[c++];
    f.cb = (const float*)din[c++]; f.cw = (const float*)din[c++];
    if (f.cin != f.cout) { f.rb = (const float*)din[c++]; f.rw = (const float*)din[c++]; }
    else { f.rb = nullptr; f.rw = nullptr; }
  }
}

extern "C" void kernel_launch(void* const* d_in, const int* in_sizes, int n_in,
                              void* d_out, int out_size, void* d_ws, size_t ws_size,
                              hipStream_t stream) {
  (void)in_sizes; (void)n_in; (void)out_size; (void)ws_size;
  // -------- parse params (alphabetical top-level keys) --------
  int c = 0;
  AttnL cross_ogp[3], cross_scene[3], self_attn[3];
  Film ogp_cond[3], out_layer[3], sample_cond[3], sample_enc[3], scene_cond[3],
      self_cond[3];
  parse_attn(d_in, c, cross_ogp);                    // cross_ogp
  parse_attn(d_in, c, cross_scene);                  // cross_scene
  parse_fcrb(d_in, c, ogp_cond, 2 * EE, EE);         // ogp_cond
  const float* ogp_l1b = (const float*)d_in[c++]; const float* ogp_l1w = (const float*)d_in[c++];
  const float* ogp_l2b = (const float*)d_in[c++]; const float* ogp_l2w = (const float*)d_in[c++];
  parse_fcrb(d_in, c, out_layer, EE, 3);             // out_layer
  parse_fcrb(d_in, c, sample_cond, 2 * EE, EE);      // sample_cond
  parse_fcrb(d_in, c, sample_enc, 3, EE);            // sample_enc
  parse_fcrb(d_in, c, scene_cond, EE + FEATC, EE);   // scene_cond
  parse_attn(d_in, c, self_attn);                    // self_attn
  parse_fcrb(d_in, c, self_cond, EE, EE);            // self_cond
  const float* time_l1b = (const float*)d_in[c++]; const float* time_l1w = (const float*)d_in[c++];
  const float* time_l2b = (const float*)d_in[c++]; const float* time_l2w = (const float*)d_in[c++];
  // -------- data inputs --------
  const float* sample = (const float*)d_in[c++];
  const int* timestep = (const int*)d_in[c++];
  const float* gcond  = (const float*)d_in[c++];
  const float* ogp    = (const float*)d_in[c++];
  const float* scnp   = (const float*)d_in[c++];
  const float* scnf   = (const float*)d_in[c++];

  // -------- workspace layout --------
  float* w = (float*)d_ws;
  size_t off = 0;
  auto alloc = [&](size_t n) { float* p = w + off; off += n; return p; };
  const size_t SLOT = (size_t)BB * 480 * 4096;
  float* s0 = alloc(SLOT);
  float* s1 = alloc(SLOT);
  float* s2 = alloc(SLOT);
  float* S  = s0;  // score matrix overlays conv slots (dead during attention)
  float* xin  = alloc((size_t)BB * 240 * 4096);
  float* xout = alloc((size_t)BB * EE * 4096);
  float* scnF = alloc((size_t)L_SCN * BB * EE);
  float* kk   = alloc((size_t)L_SCN * BB * EE);
  float* vv   = alloc((size_t)L_SCN * BB * EE);
  float* ogpF = alloc((size_t)L_OGP * BB * EE);
  float* xq   = alloc((size_t)L_SP * BB * EE);
  float* xq2  = alloc((size_t)L_SP * BB * EE);
  float* xv   = alloc((size_t)L_SP * BB * EE);
  float* qq   = alloc((size_t)L_SP * BB * EE);
  float* oo   = alloc((size_t)L_SP * BB * EE);
  float* x1   = alloc((size_t)L_SP * BB * EE);
  float* f1   = alloc((size_t)L_SP * BB * 480);
  float* f2   = alloc((size_t)L_SP * BB * EE);
  float* spe  = alloc((size_t)BB * EE * L_SP);
  float* peSC = alloc((size_t)BB * L_SCN * EE * 2);
  float* peSP = alloc((size_t)BB * L_SP * EE * 2);
  float* peOG = alloc((size_t)BB * L_OGP * EE * 2);
  float* mishg = alloc(BB * CONDD);
  float* tsin  = alloc(BB * EE);
  float* th    = alloc(BB * 480);
  float* temb  = alloc(BB * EE);
  float* obs   = alloc(NOBS * EE);
  float* act   = alloc(TT * EE);
  float* ogph  = alloc(32 * 480);
  float* ogptk = alloc(32 * EE);
  float* ebuf  = alloc(BB * 960);
  float* stat  = alloc(2 * BB);

  // -------- global conditioning --------
  k_sinpos<<<ceil_div(BB * EE, 128), 128, 0, stream>>>(timestep, 0, BB, tsin);
  gemm_tok(stream, tsin, time_l1w, time_l1b, th, BB, 480, EE, 1.f, 1);
  gemm_tok(stream, th, time_l2w, time_l2b, temb, BB, EE, 480);
  k_gfeat<<<ceil_div(BB * CONDD, 256), 256, 0, stream>>>(temb, gcond, mishg);
  k_sinpos<<<ceil_div(NOBS * EE, 128), 128, 0, stream>>>(nullptr, 1, NOBS, obs);
  k_sinpos<<<ceil_div(TT * EE, 128), 128, 0, stream>>>(nullptr, 1, TT, act);

  // -------- gripper-point stream --------
  gemm_tok(stream, ogp, ogp_l1w, ogp_l1b, ogph, 32, 480, 3, 1.f, 1);
  gemm_tok(stream, ogph, ogp_l2w, ogp_l2b, ogptk, 32, EE, 480);
  k_build_ogp_x<<<ceil_div(BB * 240 * L_OGP, 256), 256, 0, stream>>>(ogptk, obs, xin);
  run_fcrb(stream, ogp_cond, xin, xout, s0, s1, s2, mishg, ebuf, stat, L_OGP);
  k_cl_to_lbc<<<ceil_div(BB * EE * L_OGP, 256), 256, 0, stream>>>(xout, ogpF, EE, L_OGP);

  // -------- scene stream (dominant compute) --------
  k_build_scn_x<<<ceil_div(BB * 180 * L_SCN, 256), 256, 0, stream>>>(scnf, obs, xin);
  run_fcrb(stream, scene_cond, xin, xout, s0, s1, s2, mishg, ebuf, stat, L_SCN);
  k_cl_to_lbc<<<ceil_div(BB * EE * L_SCN, 256), 256, 0, stream>>>(xout, scnF, EE, L_SCN);

  // -------- sample stream --------
  k_build_sp_x<<<ceil_div(BB * 3 * L_SP, 256), 256, 0, stream>>>(sample, xin);
  run_fcrb(stream, sample_enc, xin, spe, s0, s1, s2, mishg, ebuf, stat, L_SP);
  k_build_spcat<<<ceil_div(BB * 240 * L_SP, 256), 256, 0, stream>>>(spe, act, xin);
  run_fcrb(stream, sample_cond, xin, xout, s0, s1, s2, mishg, ebuf, stat, L_SP);
  k_cl_to_lbc<<<ceil_div(BB * EE * L_SP, 256), 256, 0, stream>>>(xout, xq, EE, L_SP);

  // -------- rotary position encodings --------
  k_rope<<<ceil_div(BB * L_SP * 60, 256), 256, 0, stream>>>(sample, peSP, L_SP);
  k_rope<<<ceil_div(BB * L_OGP * 60, 256), 256, 0, stream>>>(ogp, peOG, L_OGP);
  k_rope<<<ceil_div(BB * L_SCN * 60, 256), 256, 0, stream>>>(scnp, peSC, L_SCN);

  // -------- attention stacks --------
  float* qa = xq;
  float* qb = xq2;
  for (int i = 0; i < 3; ++i) {
    run_attn(stream, cross_ogp[i], qa, ogpF, peSP, peOG, L_SP, L_OGP, qq, kk, vv, oo,
             x1, f1, f2, S, qb);
    float* t = qa; qa = qb; qb = t;
  }
  for (int i = 0; i < 3; ++i) {
    run_attn(stream, cross_scene[i], qa, scnF, peSP, peSC, L_SP, L_SCN, qq, kk, vv, oo,
             x1, f1, f2, S, qb);
    float* t = qa; qa = qb; qb = t;
  }
  // self-cond FiLM stack
  k_lbc_to_cl<<<ceil_div(BB * EE * L_SP, 256), 256, 0, stream>>>(qa, xin, EE, L_SP);
  run_fcrb(stream, self_cond, xin, xout, s0, s1, s2, mishg, ebuf, stat, L_SP);
  k_cl_to_lbc<<<ceil_div(BB * EE * L_SP, 256), 256, 0, stream>>>(xout, qa, EE, L_SP);
  // self attention (v is the fixed pre-stack x)
  (void)hipMemcpyAsync(xv, qa, sizeof(float) * (size_t)L_SP * BB * EE,
                       hipMemcpyDeviceToDevice, stream);
  for (int i = 0; i < 3; ++i) {
    run_attn(stream, self_attn[i], qa, xv, peSP, peSP, L_SP, L_SP, qq, kk, vv, oo, x1,
             f1, f2, S, qb);
    float* t = qa; qa = qb; qb = t;
  }

  // -------- output head --------
  k_lbc_to_cl<<<ceil_div(BB * EE * L_SP, 256), 256, 0, stream>>>(qa, xin, EE, L_SP);
  run_fcrb(stream, out_layer, xin, xout, s0, s1, s2, mishg, ebuf, stat, L_SP);
  k_out_final<<<ceil_div(BB * TT * 12, 256), 256, 0, stream>>>(xout, (float*)d_out);
}
